// AveragedKeyCircularConvolutionalAttention_17549236371867
// MI455X (gfx1250) — compile-verified, matrix operand order checked by reading, not run on hardware
//
#include <hip/hip_runtime.h>

#define BB 2
#define NN 2048
#define CC 1024
#define HH 16
#define DD 64

typedef __bf16 bf16;
typedef bf16 v16bf __attribute__((ext_vector_type(16)));
typedef float v8f __attribute__((ext_vector_type(8)));

union FragAB { v16bf v; uint4 q2[2]; bf16 h[16]; unsigned u[8]; };
union FragC  { v8f  v; float f[8]; };

#ifndef __has_builtin
#define __has_builtin(x) 0
#endif
#if defined(__HIP_DEVICE_COMPILE__) && __has_builtin(__builtin_amdgcn_global_load_async_to_lds_b128)
#define ASYNC_LDS 1
#else
#define ASYNC_LDS 0
#endif

#if ASYNC_LDS
typedef int v4i_ __attribute__((vector_size(16)));
typedef __attribute__((address_space(1))) v4i_* gv4p;  // global int4*
typedef __attribute__((address_space(3))) v4i_* lv4p;  // LDS int4*
#endif

// copy 16B global -> LDS; async (ASYNCcnt-tracked) when the gfx1250 builtin exists
__device__ __forceinline__ void cp_async16(const bf16* g, bf16* l) {
#if ASYNC_LDS
  __builtin_amdgcn_global_load_async_to_lds_b128(
      (gv4p)(unsigned long long)g, (lv4p)(unsigned long long)l, 0, 0);
#else
  *(uint4*)l = *(const uint4*)g;
#endif
}

__device__ __forceinline__ void wait_async_lds() {
#if ASYNC_LDS
#if __has_builtin(__builtin_amdgcn_s_wait_asynccnt)
  __builtin_amdgcn_s_wait_asynccnt(0);
#else
  asm volatile("s_wait_asynccnt 0x0" ::: "memory");
#endif
#endif
}

// CDNA5 16-bit A/B operand element->K mapping (ISA 7.12.2):
// lanes 0-15: elems 0-7 -> K 0-7, elems 8-15 -> K 16-23 ; lanes 16-31: +8
__device__ __forceinline__ int kelem(int e, int lane) {
  return ((e < 8) ? e : (e + 8)) + ((lane >= 16) ? 8 : 0);
}

__global__ void cvt_f32_bf16(const float* __restrict__ s, bf16* __restrict__ d, int n) {
  int i = blockIdx.x * blockDim.x + threadIdx.x;
  int stride = gridDim.x * blockDim.x;
  for (; i < n; i += stride) d[i] = (bf16)s[i];
}

__global__ void zero_f32(float* __restrict__ p, int n) {
  int i = blockIdx.x * blockDim.x + threadIdx.x;
  if (i < n) p[i] = 0.0f;
}

// out = A(4096 x 1024, bf16) @ W^T (W 1024x1024 row-major bf16)
// Block: 128 rows x 64 cols, 8 waves, 4 WMMA accumulators/wave.
// mode 0: outf = acc (q)        mode 1: kavg[b,c] += row-mean (k, fused)
// mode 2: outb = bf16(acc) (v)  mode 3: outf = acc + bias (projection)
__global__ __launch_bounds__(256) void gemm_bf16(
    const bf16* __restrict__ A, const bf16* __restrict__ W,
    float* __restrict__ outf, bf16* __restrict__ outb,
    float* __restrict__ kavg, const float* __restrict__ bias, int mode) {
  __shared__ bf16 wtile[2][64 * 32];  // [buf][n][k], 4KB each
  __shared__ float red[8][32];

  const int tid  = threadIdx.x;
  const int lane = tid & 31;
  const int wave = tid >> 5;
  const int c0   = blockIdx.y * 64;
  const int rblk = blockIdx.x * 128;
  const int rw   = rblk + wave * 16;
  const int arow = rw + (lane & 15);
  const int ncol = lane & 15;
  const int koff = (lane >= 16) ? 8 : 0;

  // staging: one 16B chunk per thread (64 n-rows x 4 chunks of 8 bf16)
  const int sn = tid >> 2;
  const int sk = (tid & 3) * 8;
  const bf16* wsrc = W + (size_t)(c0 + sn) * CC + sk;

  FragC acc[4];
#pragma unroll
  for (int t = 0; t < 4; t++)
#pragma unroll
    for (int r = 0; r < 8; r++) acc[t].f[r] = 0.0f;

  cp_async16(wsrc, &wtile[0][sn * 32 + sk]);
  wait_async_lds();
  __syncthreads();

  for (int kk = 0; kk < CC / 32; kk++) {
    const int k0 = kk * 32;
    const int cur = kk & 1;
    if (kk + 1 < CC / 32)
      cp_async16(wsrc + k0 + 32, &wtile[cur ^ 1][sn * 32 + sk]);

    if (k0 + 32 < CC)
      __builtin_prefetch(&A[(size_t)arow * CC + k0 + 32], 0, 1);

    FragAB fa;  // one A fragment feeds 4 WMMAs
    const uint4* ap = (const uint4*)(A + (size_t)arow * CC + k0 + koff);
    fa.q2[0] = ap[0];   // K koff..koff+7
    fa.q2[1] = ap[2];   // K koff+16..koff+23

#pragma unroll
    for (int t = 0; t < 4; t++) {
      FragAB fb;
      const uint4* bp = (const uint4*)&wtile[cur][(t * 16 + ncol) * 32 + koff];
      fb.q2[0] = bp[0];
      fb.q2[1] = bp[2];
      acc[t].v = __builtin_amdgcn_wmma_f32_16x16x32_bf16(
          false, fa.v, false, fb.v, (short)0, acc[t].v, false, false);
    }
    wait_async_lds();
    __syncthreads();
  }

  if (mode == 1) {
    const int b = rblk / NN;
#pragma unroll
    for (int t = 0; t < 4; t++) {
      float s = 0.0f;
#pragma unroll
      for (int r = 0; r < 8; r++) s += acc[t].f[r];
      red[wave][lane] = s;
      __syncthreads();
      if (tid < 16) {
        float tt = 0.0f;
#pragma unroll
        for (int w = 0; w < 8; w++) tt += red[w][tid] + red[w][tid + 16];
        atomicAdd(&kavg[b * CC + c0 + t * 16 + tid], tt * (1.0f / NN));
      }
      __syncthreads();
    }
  } else if (mode == 2) {
#pragma unroll
    for (int t = 0; t < 4; t++)
#pragma unroll
      for (int r = 0; r < 8; r++) {
        int gr = rw + r + 8 * (lane >> 4);
        outb[(size_t)gr * CC + c0 + t * 16 + ncol] = (bf16)acc[t].f[r];
      }
  } else {
#pragma unroll
    for (int t = 0; t < 4; t++) {
      float bv = (mode == 3) ? bias[c0 + t * 16 + ncol] : 0.0f;
#pragma unroll
      for (int r = 0; r < 8; r++) {
        int gr = rw + r + 8 * (lane >> 4);
        outf[(size_t)gr * CC + c0 + t * 16 + ncol] = acc[t].f[r] + bv;
      }
    }
  }
}

// per (b,h): z[n] = scale * q . kavg, softmax over n, emit bf16 attn
__global__ __launch_bounds__(256) void zsoftmax(
    const float* __restrict__ q, const float* __restrict__ kavg,
    bf16* __restrict__ attnb) {
  __shared__ float zbuf[NN];
  __shared__ float red[256];
  __shared__ float ks[DD];
  const int tid = threadIdx.x;
  const int bh = blockIdx.x;
  const int b = bh / HH, h = bh % HH;
  if (tid < DD) ks[tid] = kavg[b * CC + h * DD + tid];
  __syncthreads();
  const float scale = 0.125f;  // 64^-0.5
  const float* qb = q + (size_t)b * NN * CC + h * DD;
  for (int n = tid; n < NN; n += 256) {
    const float* qr = qb + (size_t)n * CC;
    float s = 0.0f;
#pragma unroll 8
    for (int d = 0; d < DD; d++) s += qr[d] * ks[d];
    zbuf[n] = s * scale;
  }
  __syncthreads();
  float m = -1e30f;
  for (int n = tid; n < NN; n += 256) m = fmaxf(m, zbuf[n]);
  red[tid] = m;
  __syncthreads();
  for (int s = 128; s > 0; s >>= 1) {
    if (tid < s) red[tid] = fmaxf(red[tid], red[tid + s]);
    __syncthreads();
  }
  m = red[0];
  __syncthreads();
  float ls = 0.0f;
  for (int n = tid; n < NN; n += 256) {
    float e = __expf(zbuf[n] - m);
    zbuf[n] = e;
    ls += e;
  }
  red[tid] = ls;
  __syncthreads();
  for (int s = 128; s > 0; s >>= 1) {
    if (tid < s) red[tid] += red[tid + s];
    __syncthreads();
  }
  float inv = 1.0f / red[0];
  bf16* ab = attnb + (size_t)bh * NN;
  for (int n = tid; n < NN; n += 256) ab[n] = (bf16)(zbuf[n] * inv);
}

// y[b,i,h*64+d] = sum_j attn[b,h,(j-i)%N] * v[b,j,h*64+d]
// Block: 128 i-rows x all 64 d-cols; circulant A generated from 4KB LDS attn,
// V tiles double-buffered in LDS via async copy.
__global__ __launch_bounds__(256) void circconv(
    const bf16* __restrict__ attnb, const bf16* __restrict__ v,
    bf16* __restrict__ y) {
  __shared__ bf16 asmem[NN];           // 4KB attn row
  __shared__ bf16 vtile[2][32 * 64];   // [buf][k][n], 4KB each
  const int tid  = threadIdx.x;
  const int lane = tid & 31;
  const int wave = tid >> 5;
  const int bh = blockIdx.y;
  const int b = bh / HH, h = bh % HH;
  const int i0 = blockIdx.x * 128 + wave * 16;
  const int irow = i0 + (lane & 15);
  const int ncol = lane & 15;

  const bf16* arow = attnb + (size_t)bh * NN;
  cp_async16(arow + tid * 8, &asmem[tid * 8]);  // 256 x 16B = 4KB

  // v staging: one 16B chunk per thread (32 k-rows x 8 chunks of 8 bf16)
  const int sk = tid >> 3;
  const int sn = (tid & 7) * 8;
  const bf16* vsrc = v + (size_t)(b * NN + sk) * CC + h * DD + sn;

  cp_async16(vsrc, &vtile[0][sk * 64 + sn]);
  wait_async_lds();
  __syncthreads();

  FragC acc[4];
#pragma unroll
  for (int t = 0; t < 4; t++)
#pragma unroll
    for (int r = 0; r < 8; r++) acc[t].f[r] = 0.0f;

  for (int jj = 0; jj < NN / 32; jj++) {
    const int j0 = jj * 32;
    const int cur = jj & 1;
    if (jj + 1 < NN / 32)
      cp_async16(vsrc + (size_t)(j0 + 32) * CC, &vtile[cur ^ 1][sk * 64 + sn]);

    FragAB fa;  // circulant gather R[i,j] = attn[(j-i) mod N]; feeds 4 WMMAs
#pragma unroll
    for (int e = 0; e < 16; e++) {
      int j = j0 + kelem(e, lane);
      fa.h[e] = asmem[(j - irow) & (NN - 1)];
    }
#pragma unroll
    for (int t = 0; t < 4; t++) {
      FragAB fb;
#pragma unroll
      for (int e = 0; e < 16; e++)
        fb.h[e] = vtile[cur][kelem(e, lane) * 64 + t * 16 + ncol];
      acc[t].v = __builtin_amdgcn_wmma_f32_16x16x32_bf16(
          false, fa.v, false, fb.v, (short)0, acc[t].v, false, false);
    }
    wait_async_lds();
    __syncthreads();
  }
#pragma unroll
  for (int t = 0; t < 4; t++)
#pragma unroll
    for (int r = 0; r < 8; r++) {
      int gi = i0 + r + 8 * (lane >> 4);
      y[(size_t)(b * NN + gi) * CC + h * DD + t * 16 + ncol] = (bf16)acc[t].f[r];
    }
}

extern "C" void kernel_launch(void* const* d_in, const int* in_sizes, int n_in,
                              void* d_out, int out_size, void* d_ws, size_t ws_size,
                              hipStream_t stream) {
  const float* x  = (const float*)d_in[0];
  const float* wq = (const float*)d_in[1];
  const float* wk = (const float*)d_in[2];
  const float* wv = (const float*)d_in[3];
  const float* wp = (const float*)d_in[4];
  const float* bp = (const float*)d_in[5];
  float* out = (float*)d_out;

  char* ws = (char*)d_ws;
  size_t off = 0;
  auto alloc = [&](size_t bytes) -> void* {
    void* p = ws + off;
    off = (off + bytes + 255) & ~(size_t)255;
    return p;
  };

  bf16*  xb    = (bf16*)alloc((size_t)BB * NN * CC * 2);
  bf16*  wqb   = (bf16*)alloc((size_t)CC * CC * 2);
  bf16*  wkb   = (bf16*)alloc((size_t)CC * CC * 2);
  bf16*  wvb   = (bf16*)alloc((size_t)CC * CC * 2);
  bf16*  wpb   = (bf16*)alloc((size_t)CC * CC * 2);
  float* qf    = (float*)alloc((size_t)BB * NN * CC * 4);
  bf16*  vb    = (bf16*)alloc((size_t)BB * NN * CC * 2);
  float* kavg  = (float*)alloc((size_t)BB * CC * 4);
  bf16*  attnb = (bf16*)alloc((size_t)BB * HH * NN * 2);
  bf16*  yb    = (bf16*)alloc((size_t)BB * NN * CC * 2);

  cvt_f32_bf16<<<2048, 256, 0, stream>>>(x, xb, BB * NN * CC);
  cvt_f32_bf16<<<1024, 256, 0, stream>>>(wq, wqb, CC * CC);
  cvt_f32_bf16<<<1024, 256, 0, stream>>>(wk, wkb, CC * CC);
  cvt_f32_bf16<<<1024, 256, 0, stream>>>(wv, wvb, CC * CC);
  cvt_f32_bf16<<<1024, 256, 0, stream>>>(wp, wpb, CC * CC);
  zero_f32<<<(BB * CC + 255) / 256, 256, 0, stream>>>(kavg, BB * CC);

  dim3 gg(BB * NN / 128, CC / 64);  // (32, 16)
  gemm_bf16<<<gg, 256, 0, stream>>>(xb, wqb, qf, nullptr, nullptr, nullptr, 0);
  gemm_bf16<<<gg, 256, 0, stream>>>(xb, wkb, nullptr, nullptr, kavg, nullptr, 1);
  gemm_bf16<<<gg, 256, 0, stream>>>(xb, wvb, nullptr, vb, nullptr, nullptr, 2);

  zsoftmax<<<BB * HH, 256, 0, stream>>>(qf, kavg, attnb);

  dim3 gc(NN / 128, BB * HH);  // (16, 32)
  circconv<<<gc, 256, 0, stream>>>(attnb, vb, yb);

  gemm_bf16<<<gg, 256, 0, stream>>>(yb, wpb, out, nullptr, nullptr, bp, 3);
}
